// QuantizedLinear_80453327389436
// MI455X (gfx1250) — compile-verified, matrix operand order checked by reading, not run on hardware
//
#include <hip/hip_runtime.h>
#include <stdint.h>

typedef _Float16 v8h  __attribute__((ext_vector_type(8)));
typedef _Float16 v16h __attribute__((ext_vector_type(16)));
typedef _Float16 h2   __attribute__((ext_vector_type(2)));
typedef float    v8f  __attribute__((ext_vector_type(8)));

#define K_DIM 4096
#define N_DIM 16384
#define BM 128
#define BN 256
#define BK 32
#define LDA 40   // halves per LDS row: 32 + 8 pad (spreads the 64 banks)

__global__ __launch_bounds__(256)
void qlinear_wmma_kernel(const _Float16* __restrict__ x,
                         const int8_t*   __restrict__ wq,
                         const _Float16* __restrict__ scales,
                         const _Float16* __restrict__ bias,
                         _Float16*       __restrict__ out)
{
    __shared__ _Float16 As[2][BM * LDA];   // 20 KB
    __shared__ _Float16 Bs[2][BN * LDA];   // 40 KB

    const int tid = threadIdx.x;
    const int bm0 = blockIdx.y * BM;
    const int bn0 = blockIdx.x * BN;

    // ---- A staging: 2 threads per row, 16 halves each ----
    const int lrow = tid >> 1;          // 0..127
    const int lkh  = (tid & 1) * 16;    // K offset in halves: 0 or 16
    const _Float16* xrow = x + (size_t)(bm0 + lrow) * K_DIM + lkh;

    // ---- B staging: 1 thread per row, full 32-byte K chunk ----
    const int8_t*  wrowB = wq + (size_t)(bn0 + tid) * K_DIM;
    const _Float16 srow  = scales[bn0 + tid];

    // magic-number dequant constants: fp16(0x6400|(b^0x80)) == 1152 + w
    h2 s2;  s2[0] = srow;  s2[1] = srow;
    const _Float16 nbv = (_Float16)(-1152.0f) * srow;
    h2 nb2; nb2[0] = nbv;  nb2[1] = nbv;

    // ---- wave / lane mapping (wave32): 8 waves in 2(M) x 4(N), 64x64 each ----
    const int lane = tid & 31;
    const int wid  = tid >> 5;
    const int wm   = (wid & 1) * 64;
    const int wn   = (wid >> 1) * 64;
    const int lr   = lane & 15;
    const int kh8  = (lane >> 4) * 8;

    v8f acc[4][4] = {};

    // A tile: pure copy -> CDNA5 async global->LDS DMA (ASYNCcnt-tracked).
    auto stageA_async = [&](int buf, int ko) {
        const uint32_t ldsa = (uint32_t)(uintptr_t)&As[buf][lrow * LDA + lkh];
        const uint64_t ga   = (uint64_t)(uintptr_t)(xrow + ko);
        asm volatile(
            "global_load_async_to_lds_b128 %0, %1, off\n\t"
            "global_load_async_to_lds_b128 %0, %1, off offset:16"
            :: "v"(ldsa), "v"(ga) : "memory");
    };

    // B tile: int8 -> fp16*scale via xor + v_perm + v_pk_fma_f16 (32 bytes/thread)
    auto stageB = [&](int buf, int4 b0, int4 b1) {
        union { int4 v[2]; uint32_t u[8]; } bw; bw.v[0] = b0; bw.v[1] = b1;
        uint32_t ow[16];
        #pragma unroll
        for (int d = 0; d < 8; ++d) {
            const uint32_t u = bw.u[d] ^ 0x80808080u;            // bias to [0,255]
            uint32_t plo = __builtin_amdgcn_perm(0x64646464u, u, 0x04010400u);
            uint32_t phi = __builtin_amdgcn_perm(0x64646464u, u, 0x04030402u);
            h2 hlo = __builtin_bit_cast(h2, plo);                // = 1152 + w
            h2 hhi = __builtin_bit_cast(h2, phi);
            hlo = __builtin_elementwise_fma(hlo, s2, nb2);       // = w * s
            hhi = __builtin_elementwise_fma(hhi, s2, nb2);
            ow[2*d]   = __builtin_bit_cast(uint32_t, hlo);
            ow[2*d+1] = __builtin_bit_cast(uint32_t, hhi);
        }
        #pragma unroll
        for (int q = 0; q < 4; ++q) {
            uint4 qq;
            qq.x = ow[4*q]; qq.y = ow[4*q+1]; qq.z = ow[4*q+2]; qq.w = ow[4*q+3];
            *(uint4*)&Bs[buf][tid * LDA + q * 8] = qq;
        }
    };

    // prologue: tile 0 -> buffer 0
    stageA_async(0, 0);
    {
        int4 b0 = *(const int4*)(wrowB);
        int4 b1 = *(const int4*)(wrowB + 16);
        stageB(0, b0, b1);
    }
    asm volatile("s_wait_asynccnt 0x0" ::: "memory");
    __syncthreads();

    const int NK = K_DIM / BK;   // 128 K-steps
    for (int kt = 0; kt < NK; ++kt) {
        const int cur = kt & 1;

        // issue next tile's transfers early to overlap with WMMA work
        int4 b0 = {}, b1 = {};
        if (kt + 1 < NK) {
            const int ko = (kt + 1) * BK;
            stageA_async(cur ^ 1, ko);
            b0 = *(const int4*)(wrowB + ko);
            b1 = *(const int4*)(wrowB + ko + 16);
        }

        // fragment loads per ISA 7.12.2 layout: two b128 per 16x32 fragment
        v16h afrag[4], bfrag[4];
        #pragma unroll
        for (int i = 0; i < 4; ++i) {
            const _Float16* p = &As[cur][(wm + i * 16 + lr) * LDA + kh8];
            v8h lo = *(const v8h*)p;
            v8h hi = *(const v8h*)(p + 16);
            afrag[i] = __builtin_shufflevector(lo, hi,
                0,1,2,3,4,5,6,7,8,9,10,11,12,13,14,15);
        }
        #pragma unroll
        for (int j = 0; j < 4; ++j) {
            const _Float16* p = &Bs[cur][(wn + j * 16 + lr) * LDA + kh8];
            v8h lo = *(const v8h*)p;
            v8h hi = *(const v8h*)(p + 16);
            bfrag[j] = __builtin_shufflevector(lo, hi,
                0,1,2,3,4,5,6,7,8,9,10,11,12,13,14,15);
        }

        // 16x v_wmma_f32_16x16x32_f16 per wave per K-step
        #pragma unroll
        for (int i = 0; i < 4; ++i)
            #pragma unroll
            for (int j = 0; j < 4; ++j)
                acc[i][j] = __builtin_amdgcn_wmma_f32_16x16x32_f16(
                    false, afrag[i], false, bfrag[j],
                    (short)0, acc[i][j], false, false);

        if (kt + 1 < NK) stageB(cur ^ 1, b0, b1);
        asm volatile("s_wait_asynccnt 0x0" ::: "memory");
        __syncthreads();
    }

    // ---- epilogue: C layout (lane = N + 16*(M>=8), vgpr r = M%8), add bias ----
    const int m0 = bm0 + wm + (lane >> 4) * 8;
    const int n0 = bn0 + wn + lr;
    #pragma unroll
    for (int j = 0; j < 4; ++j) {
        const float bf = (float)bias[n0 + j * 16];
        #pragma unroll
        for (int i = 0; i < 4; ++i) {
            #pragma unroll
            for (int r = 0; r < 8; ++r) {
                out[(size_t)(m0 + i * 16 + r) * N_DIM + (n0 + j * 16)] =
                    (_Float16)(acc[i][j][r] + bf);
            }
        }
    }
}

extern "C" void kernel_launch(void* const* d_in, const int* in_sizes, int n_in,
                              void* d_out, int out_size, void* d_ws, size_t ws_size,
                              hipStream_t stream) {
    const _Float16* x  = (const _Float16*)d_in[0];   // [M, 4096] fp16
    const int8_t*   wq = (const int8_t*)  d_in[1];   // [16384, 4096] int8
    const _Float16* sc = (const _Float16*)d_in[2];   // [16384] fp16
    const _Float16* bi = (const _Float16*)d_in[3];   // [16384] fp16
    _Float16*       y  = (_Float16*)d_out;           // [M, 16384] fp16

    const int M = in_sizes[0] / K_DIM;               // 8192
    dim3 grid(N_DIM / BN, M / BM);                   // (64, 64)
    qlinear_wmma_kernel<<<grid, dim3(256), 0, stream>>>(x, wq, sc, bi, y);
}
